// VisionTransformer_52905407152495
// MI455X (gfx1250) — matrix operand order repaired and checked by reading, not compiled, not run on hardware
//
#include <hip/hip_runtime.h>
#include <hip/hip_bf16.h>

// ---------------------------------------------------------------------------
// ViT-B/16 forward for MI455X (gfx1250).
//  - bf16 WMMA GEMM core (v_wmma_f32_16x16x32_bf16, f32 accum)
//  - 128x128 block tile, 8 waves x (32x64), double-buffered LDS pipeline
//  - B operand bf16 [N][K], staged with global_load_async_to_lds_b128 (ASYNCcnt)
//  - A operand fp32 -> bf16 at staging (float4 loads, branch-free guards)
// ---------------------------------------------------------------------------

typedef __attribute__((ext_vector_type(16))) __bf16 v16bf;
typedef __attribute__((ext_vector_type(8)))  __bf16 v8bf;
typedef __attribute__((ext_vector_type(4)))  __bf16 v4bf;
typedef __attribute__((ext_vector_type(8)))  float  v8f;

#define VIT_B   32
#define VIT_S   197
#define VIT_SP  208
#define VIT_D   768
#define VIT_FF  3072
#define VIT_L   12
#define VIT_NH  12
#define VIT_HD  64
#define VIT_NC  1000
#define VIT_NP  196
#define NROWS   (VIT_B * VIT_S)   // 6304

#define BM   128
#define BN   128
#define BKT  64
#define APAD 8
#define LDSB (BKT + APAD)   // 72 shorts = 144 B (multiple of 16 B)

// ---------------- staging helpers ----------------

// Load A tile (BM x BKT fp32) into registers. Unified va[32] for both paths.
__device__ __forceinline__ void loadA_fast(const float* __restrict__ Ap, int lda,
                                           int bm0, int k0, int tid, float* va)
{
#pragma unroll
    for (int i = 0; i < 8; ++i) {
        int e  = tid + i * 256;          // 2048 float4 slots
        int m  = e >> 4;
        int kq = (e & 15) << 2;
        const float4 f = *(const float4*)(Ap + (long long)(bm0 + m) * lda + (k0 + kq));
        va[4 * i + 0] = f.x; va[4 * i + 1] = f.y; va[4 * i + 2] = f.z; va[4 * i + 3] = f.w;
    }
}

__device__ __forceinline__ void storeA_fast(__bf16 (*as)[LDSB], int tid, const float* va)
{
#pragma unroll
    for (int i = 0; i < 8; ++i) {
        int e  = tid + i * 256;
        int m  = e >> 4;
        int kq = (e & 15) << 2;
        v4bf o;
        o[0] = (__bf16)va[4 * i + 0]; o[1] = (__bf16)va[4 * i + 1];
        o[2] = (__bf16)va[4 * i + 2]; o[3] = (__bf16)va[4 * i + 3];
        *(v4bf*)&as[m][kq] = o;
    }
}

__device__ __forceinline__ void loadA_guard(const float* __restrict__ Ap, int lda,
                                            int bm0, int k0, int M, int K, int tid, float* va)
{
#pragma unroll
    for (int i = 0; i < 32; ++i) {
        int e  = tid + i * 256;
        int m  = e >> 6, kk = e & 63;
        int gm = bm0 + m, gk = k0 + kk;
        int cm = gm < M ? gm : M - 1;
        int ck = gk < K ? gk : K - 1;
        float v = Ap[(long long)cm * lda + ck];
        va[i] = (gm < M && gk < K) ? v : 0.0f;
    }
}

__device__ __forceinline__ void storeA_guard(__bf16 (*as)[LDSB], int tid, const float* va)
{
#pragma unroll
    for (int i = 0; i < 32; ++i) {
        int e = tid + i * 256;
        as[e >> 6][e & 63] = (__bf16)va[i];
    }
}

// Stage B tile (BN x BKT bf16 raw copy) into LDS buffer.
__device__ __forceinline__ void stageB(const __bf16* __restrict__ Bp, int ldb,
                                       int bn0, int k0, int N, int K, bool fast,
                                       int tid, __bf16 (*bs)[LDSB])
{
    if (fast) {
        // async global->LDS: 1024 x 16-byte chunks, 4 per thread
#pragma unroll
        for (int i = 0; i < 4; ++i) {
            int e = tid + i * 256;
            int n = e >> 3;
            int c = (e & 7) << 3;
            unsigned ldsaddr = (unsigned)(unsigned long long)&bs[n][c];
            const __bf16* gp = Bp + (long long)(bn0 + n) * ldb + (k0 + c);
            asm volatile("global_load_async_to_lds_b128 %0, %1, off"
                         :: "v"(ldsaddr), "v"(gp) : "memory");
        }
    } else {
#pragma unroll
        for (int i = 0; i < 32; ++i) {
            int e  = tid + i * 256;
            int n  = e >> 6, kk = e & 63;
            int gn = bn0 + n, gk = k0 + kk;
            int cn = gn < N ? gn : N - 1;
            int ck = gk < K ? gk : K - 1;
            __bf16 v = Bp[(long long)cn * ldb + ck];
            bs[n][kk] = (gn < N && gk < K) ? v : (__bf16)0.0f;
        }
    }
}

// ---------------------------------------------------------------------------
// GEMM:  C = act(A @ Bt^T + bias) (+ residual)
//   A : fp32 [M][K] (lda);  Bt : bf16 [N][K] (ldb)
// Batched over grid.z: off = (z/innerCnt)*outer + (z%innerCnt)*inner.
// ---------------------------------------------------------------------------
template <int ACT, int RES>
__global__ void __launch_bounds__(256)
gemm_wmma_bf16(const float* __restrict__ A, const __bf16* __restrict__ Bt,
               const float* __restrict__ bias, const float* __restrict__ Res,
               float* __restrict__ C,
               int M, int N, int K, int lda, int ldb, int ldc,
               int innerCnt,
               long long outA, long long inA,
               long long outB, long long inB,
               long long outC, long long inC)
{
    __shared__ __attribute__((aligned(16))) __bf16 As[2][BM][LDSB];
    __shared__ __attribute__((aligned(16))) __bf16 Bs[2][BN][LDSB];

    const int tid   = threadIdx.x;
    const int lane  = tid & 31;
    const int wid   = tid >> 5;       // 0..7
    const int wm    = wid >> 1;       // 0..3  -> 32 rows each
    const int wn    = wid & 1;        // 0..1  -> 64 cols each
    const int lhalf = lane >> 4;
    const int lrow  = lane & 15;

    const int zb = blockIdx.z;
    const long long offA = (long long)(zb / innerCnt) * outA + (long long)(zb % innerCnt) * inA;
    const long long offB = (long long)(zb / innerCnt) * outB + (long long)(zb % innerCnt) * inB;
    const long long offC = (long long)(zb / innerCnt) * outC + (long long)(zb % innerCnt) * inC;

    const float*  Ap = A  + offA;
    const __bf16* Bp = Bt + offB;

    const int bm0 = blockIdx.y * BM;
    const int bn0 = blockIdx.x * BN;
    const bool fullM = (bm0 + BM) <= M;
    const bool fullN = (bn0 + BN) <= N;

    v8f acc[2][4];
#pragma unroll
    for (int i = 0; i < 2; ++i)
#pragma unroll
        for (int j = 0; j < 4; ++j)
#pragma unroll
            for (int r = 0; r < 8; ++r) acc[i][j][r] = 0.0f;

    float va[32];

    // ---------------- prologue: stage tile 0 into buffer 0 ----------------
    {
        const bool fk = BKT <= K;
        const bool fa = fullM && fk;
        if (fa) { loadA_fast(Ap, lda, bm0, 0, tid, va); storeA_fast(As[0], tid, va); }
        else    { loadA_guard(Ap, lda, bm0, 0, M, K, tid, va); storeA_guard(As[0], tid, va); }
        stageB(Bp, ldb, bn0, 0, N, K, fullN && fk, tid, Bs[0]);
        asm volatile("s_wait_asynccnt 0x0" ::: "memory");
        __syncthreads();
    }

    int buf = 0;
    for (int k0 = 0; k0 < K; k0 += BKT) {
        const int  nxt      = buf ^ 1;
        const bool has_next = (k0 + BKT) < K;
        const int  k0n      = k0 + BKT;
        const bool fkn      = (k0n + BKT) <= K;
        const bool fan      = fullM && fkn;

        // 1) issue next tile's loads (A -> regs, B -> async into LDS[nxt])
        if (has_next) {
            if (fan) loadA_fast(Ap, lda, bm0, k0n, tid, va);
            else     loadA_guard(Ap, lda, bm0, k0n, M, K, tid, va);
            stageB(Bp, ldb, bn0, k0n, N, K, fullN && fkn, tid, Bs[nxt]);
        }

        // 2) compute on current buffer
        __bf16 (*as)[LDSB] = As[buf];
        __bf16 (*bs)[LDSB] = Bs[buf];
#pragma unroll
        for (int ks = 0; ks < BKT; ks += 32) {
            v16bf afrag[2], bfrag[4];
#pragma unroll
            for (int i = 0; i < 2; ++i) {
                const __bf16* p0 = &as[wm * 32 + i * 16 + lrow][ks + lhalf * 8];
                v8bf lo = *(const v8bf*)p0;
                v8bf hi = *(const v8bf*)(p0 + 16);
#pragma unroll
                for (int t = 0; t < 8; ++t) { afrag[i][t] = lo[t]; afrag[i][8 + t] = hi[t]; }
            }
#pragma unroll
            for (int j = 0; j < 4; ++j) {
                const __bf16* p0 = &bs[wn * 64 + j * 16 + lrow][ks + lhalf * 8];
                v8bf lo = *(const v8bf*)p0;
                v8bf hi = *(const v8bf*)(p0 + 16);
#pragma unroll
                for (int t = 0; t < 8; ++t) { bfrag[j][t] = lo[t]; bfrag[j][8 + t] = hi[t]; }
            }
#pragma unroll
            for (int i = 0; i < 2; ++i)
#pragma unroll
                for (int j = 0; j < 4; ++j)
                    acc[i][j] = __builtin_amdgcn_wmma_f32_16x16x32_bf16(
                        false, afrag[i], false, bfrag[j], (short)0, acc[i][j], false, false);
        }

        // 3) convert + store next A tile into LDS[nxt]
        if (has_next) {
            if (fan) storeA_fast(As[nxt], tid, va);
            else     storeA_guard(As[nxt], tid, va);
        }

        // 4) close stage: async B done + all LDS writes visible
        asm volatile("s_wait_asynccnt 0x0" ::: "memory");
        __syncthreads();
        buf = nxt;
    }

    // ---------------- epilogue ----------------
#pragma unroll
    for (int i = 0; i < 2; ++i) {
#pragma unroll
        for (int j = 0; j < 4; ++j) {
            int gc = bn0 + wn * 64 + j * 16 + lrow;
            float bv = 0.0f;
            if (bias && gc < N) bv = bias[gc];
#pragma unroll
            for (int r = 0; r < 8; ++r) {
                int gr = bm0 + wm * 32 + i * 16 + lhalf * 8 + r;
                if (gr < M && gc < N) {
                    float v = acc[i][j][r] + bv;
                    if (ACT == 1) v = 0.5f * v * (1.0f + erff(v * 0.70710678118654752f));
                    long long ci = offC + (long long)gr * ldc + gc;
                    if (RES) v += Res[ci];
                    C[ci] = v;
                }
            }
        }
    }
}

// ---------------------------------------------------------------------------
// Tiled transpose + fp32->bf16:  in fp32 [R][Ccols] (ldi) -> out bf16 [Ccols][R] (ldo)
// ---------------------------------------------------------------------------
__global__ void __launch_bounds__(256)
transpose_cvt_k(const float* __restrict__ in, __bf16* __restrict__ out,
                int R, int Ccols, int ldi, int ldo,
                int innerCnt,
                long long outI, long long inI, long long outO, long long inO)
{
    __shared__ float t[32][33];
    const int tid = threadIdx.x;
    const int tx = tid & 31, ty = tid >> 5;
    const int zb = blockIdx.z;
    const long long offI = (long long)(zb / innerCnt) * outI + (long long)(zb % innerCnt) * inI;
    const long long offO = (long long)(zb / innerCnt) * outO + (long long)(zb % innerCnt) * inO;
    const int r0 = blockIdx.y * 32, c0 = blockIdx.x * 32;

#pragma unroll
    for (int j = 0; j < 32; j += 8) {
        int r = r0 + ty + j, c = c0 + tx;
        int cr = r < R ? r : R - 1;
        int cc = c < Ccols ? c : Ccols - 1;
        float v = in[offI + (long long)cr * ldi + cc];
        t[ty + j][tx] = (r < R && c < Ccols) ? v : 0.0f;
    }
    __syncthreads();
#pragma unroll
    for (int j = 0; j < 32; j += 8) {
        int c = c0 + ty + j;
        int r = r0 + tx;
        if (c < Ccols && r < R)
            out[offO + (long long)c * ldo + r] = (__bf16)t[tx][ty + j];
    }
}

__global__ void __launch_bounds__(256)
cvt_bf16_k(const float* __restrict__ in, __bf16* __restrict__ out, long long n)
{
    long long i = ((long long)blockIdx.x * 256 + threadIdx.x) * 4;
    if (i + 3 < n) {
        const float4 f = *(const float4*)(in + i);
        v4bf o;
        o[0] = (__bf16)f.x; o[1] = (__bf16)f.y; o[2] = (__bf16)f.z; o[3] = (__bf16)f.w;
        *(v4bf*)(out + i) = o;
    } else {
        for (; i < n; ++i) out[i] = (__bf16)in[i];
    }
}

// ---------------------------------------------------------------------------
// LayerNorm (eps OUTSIDE sqrt). One block per row, D=768.
// ---------------------------------------------------------------------------
__global__ void __launch_bounds__(256)
layernorm_k(const float* __restrict__ x, const float* __restrict__ w,
            const float* __restrict__ b, float* __restrict__ y, long long xstride)
{
    __shared__ float s1[256], s2[256];
    const int tid = threadIdx.x;
    const float* xr = x + (long long)blockIdx.x * xstride;
    float* yr = y + (long long)blockIdx.x * VIT_D;

    float v0 = xr[tid], v1 = xr[tid + 256], v2 = xr[tid + 512];
    s1[tid] = v0 + v1 + v2;
    s2[tid] = v0 * v0 + v1 * v1 + v2 * v2;
    __syncthreads();
    for (int st = 128; st > 0; st >>= 1) {
        if (tid < st) { s1[tid] += s1[tid + st]; s2[tid] += s2[tid + st]; }
        __syncthreads();
    }
    float mean = s1[0] * (1.0f / VIT_D);
    float var  = s2[0] * (1.0f / VIT_D) - mean * mean;
    var = var < 0.0f ? 0.0f : var;
    float inv = 1.0f / (sqrtf(var) + 1e-5f);
    yr[tid]       = (v0 - mean) * inv * w[tid]       + b[tid];
    yr[tid + 256] = (v1 - mean) * inv * w[tid + 256] + b[tid + 256];
    yr[tid + 512] = (v2 - mean) * inv * w[tid + 512] + b[tid + 512];
}

// ---------------------------------------------------------------------------
// Softmax per (head,query) row; 197 valid cols, scale 1/8.
// ---------------------------------------------------------------------------
__global__ void __launch_bounds__(256)
softmax_k(float* __restrict__ p)
{
    __shared__ float red[256];
    const int tid = threadIdx.x;
    const int row = blockIdx.x;
    const int g = row / VIT_S;
    const int q = row % VIT_S;
    float* pr = p + (long long)g * VIT_SP * VIT_SP + (long long)q * VIT_SP;

    float v = (tid < VIT_S) ? pr[tid] * 0.125f : -3.4e38f;
    red[tid] = v;
    __syncthreads();
    for (int st = 128; st > 0; st >>= 1) {
        if (tid < st) red[tid] = fmaxf(red[tid], red[tid + st]);
        __syncthreads();
    }
    float mx = red[0];
    __syncthreads();
    float e = (tid < VIT_S) ? expf(v - mx) : 0.0f;
    red[tid] = e;
    __syncthreads();
    for (int st = 128; st > 0; st >>= 1) {
        if (tid < st) red[tid] += red[tid + st];
        __syncthreads();
    }
    float s = red[0];
    if (tid < VIT_S) pr[tid] = e / s;
}

// ---------------------------------------------------------------------------
// im2col + sequence assembly
// ---------------------------------------------------------------------------
__global__ void __launch_bounds__(256)
im2col_k(const float* __restrict__ x, float* __restrict__ out)
{
    long long idx = (long long)blockIdx.x * 256 + threadIdx.x;
    const long long total = (long long)VIT_B * VIT_NP * VIT_D;
    if (idx >= total) return;
    int col = (int)(idx % VIT_D);
    int row = (int)(idx / VIT_D);
    int b = row / VIT_NP, p = row % VIT_NP;
    int py = p / 14, px = p % 14;
    int c = col / 256, r = col % 256;
    int iy = r / 16, ix = r % 16;
    out[idx] = x[(((long long)b * 3 + c) * 224 + (py * 16 + iy)) * 224 + (px * 16 + ix)];
}

__global__ void __launch_bounds__(256)
build_h_k(const float* __restrict__ emb, const float* __restrict__ cls,
          const float* __restrict__ pos, float* __restrict__ h)
{
    long long idx = (long long)blockIdx.x * 256 + threadIdx.x;
    const long long total = (long long)VIT_B * VIT_S * VIT_D;
    if (idx >= total) return;
    int d = (int)(idx % VIT_D);
    long long rem = idx / VIT_D;
    int s = (int)(rem % VIT_S);
    long long b = rem / VIT_S;
    float v = (s == 0) ? cls[d] : emb[((long long)b * VIT_NP + (s - 1)) * VIT_D + d];
    h[idx] = v + pos[(long long)s * VIT_D + d];
}

// ---------------------------------------------------------------------------
// Host helpers
// ---------------------------------------------------------------------------
template <int ACT, int RES>
static void launch_gemm(hipStream_t s,
                        const float* A, const __bf16* Bt, const float* bias,
                        const float* Res, float* C,
                        int M, int N, int K, int lda, int ldb, int ldc,
                        int batch, int innerCnt,
                        long long outA, long long inA,
                        long long outB, long long inB,
                        long long outC, long long inC)
{
    dim3 grid((N + BN - 1) / BN, (M + BM - 1) / BM, batch);
    gemm_wmma_bf16<ACT, RES><<<grid, dim3(256), 0, s>>>(
        A, Bt, bias, Res, C, M, N, K, lda, ldb, ldc,
        innerCnt, outA, inA, outB, inB, outC, inC);
}

static void launch_trcvt(hipStream_t s, const float* in, __bf16* out,
                         int R, int Ccols, int ldi, int ldo,
                         int batch = 1, int innerCnt = 1,
                         long long outI = 0, long long inI = 0,
                         long long outO = 0, long long inO = 0)
{
    dim3 grid((Ccols + 31) / 32, (R + 31) / 32, batch);
    transpose_cvt_k<<<grid, dim3(256), 0, s>>>(in, out, R, Ccols, ldi, ldo,
                                               innerCnt, outI, inI, outO, inO);
}

extern "C" void kernel_launch(void* const* d_in, const int* in_sizes, int n_in,
                              void* d_out, int out_size, void* d_ws, size_t ws_size,
                              hipStream_t stream)
{
    const float* x       = (const float*)d_in[0];
    const float* patch_w = (const float*)d_in[1];
    const float* patch_b = (const float*)d_in[2];
    const float* cls_tok = (const float*)d_in[3];
    const float* pos_emb = (const float*)d_in[4];
    const float* ln1_w   = (const float*)d_in[5];
    const float* ln1_b   = (const float*)d_in[6];
    const float* wq      = (const float*)d_in[7];
    const float* bq      = (const float*)d_in[8];
    const float* wk      = (const float*)d_in[9];
    const float* bk      = (const float*)d_in[10];
    const float* wv      = (const float*)d_in[11];
    const float* bv      = (const float*)d_in[12];
    const float* wy      = (const float*)d_in[13];
    const float* by      = (const float*)d_in[14];
    const float* ln2_w   = (const float*)d_in[15];
    const float* ln2_b   = (const float*)d_in[16];
    const float* m1w     = (const float*)d_in[17];
    const float* m1b     = (const float*)d_in[18];
    const float* m2w     = (const float*)d_in[19];
    const float* m2b     = (const float*)d_in[20];
    const float* hln_w   = (const float*)d_in[21];
    const float* hln_b   = (const float*)d_in[22];
    const float* head_w  = (const float*)d_in[23];
    const float* head_b  = (const float*)d_in[24];
    float* out = (float*)d_out;
    (void)in_sizes; (void)n_in; (void)out_size; (void)ws_size;

    float* ws = (float*)d_ws;
    size_t o = 0;
    float* patches = ws + o; o += (size_t)VIT_B * VIT_NP * VIT_D;
    float* emb     = ws + o; o += (size_t)VIT_B * VIT_NP * VIT_D;
    float* h       = ws + o; o += (size_t)NROWS * VIT_D;
    float* z       = ws + o; o += (size_t)NROWS * VIT_D;
    float* qb      = ws + o; o += (size_t)NROWS * VIT_D;
    float* kb      = ws + o; o += (size_t)NROWS * VIT_D;
    float* vb      = ws + o; o += (size_t)NROWS * VIT_D;
    float* yb      = ws + o; o += (size_t)NROWS * VIT_D;
    float* probs   = ws + o; o += (size_t)VIT_B * VIT_NH * VIT_SP * VIT_SP;
    float* ff      = ws + o; o += (size_t)NROWS * VIT_FF;
    float* clsn    = ws + o; o += (size_t)VIT_B * VIT_D;
    __bf16* wT  = (__bf16*)(ws + o); o += (size_t)VIT_D * VIT_FF / 2;
    __bf16* kbf = (__bf16*)(ws + o); o += (size_t)NROWS * VIT_D / 2;
    __bf16* vt  = (__bf16*)(ws + o); o += (size_t)VIT_B * VIT_NH * VIT_HD * VIT_SP / 2;

    const long long SD  = (long long)VIT_S * VIT_D;
    const long long PP  = (long long)VIT_SP * VIT_SP;
    const long long VTH = (long long)VIT_HD * VIT_SP;

    // ---- patch embedding ----
    {
        long long tot = (long long)VIT_B * VIT_NP * VIT_D;
        im2col_k<<<dim3((unsigned)((tot + 255) / 256)), dim3(256), 0, stream>>>(x, patches);
        launch_trcvt(stream, patch_w, wT, VIT_D, VIT_D, VIT_D, VIT_D);
        launch_gemm<0, 0>(stream, patches, wT, patch_b, nullptr, emb,
                          VIT_B * VIT_NP, VIT_D, VIT_D, VIT_D, VIT_D, VIT_D,
                          1, 1, 0, 0, 0, 0, 0, 0);
        tot = (long long)NROWS * VIT_D;
        build_h_k<<<dim3((unsigned)((tot + 255) / 256)), dim3(256), 0, stream>>>(emb, cls_tok, pos_emb, h);
    }

    // ---- transformer blocks ----
    for (int l = 0; l < VIT_L; ++l) {
        const float* l1w = ln1_w + (size_t)l * VIT_D;
        const float* l1b = ln1_b + (size_t)l * VIT_D;
        const float* l2w = ln2_w + (size_t)l * VIT_D;
        const float* l2b = ln2_b + (size_t)l * VIT_D;
        const float* wq_l = wq + (size_t)l * VIT_D * VIT_D;
        const float* wk_l = wk + (size_t)l * VIT_D * VIT_D;
        const float* wv_l = wv + (size_t)l * VIT_D * VIT_D;
        const float* wy_l = wy + (size_t)l * VIT_D * VIT_D;
        const float* bq_l = bq + (size_t)l * VIT_D;
        const float* bk_l = bk + (size_t)l * VIT_D;
        const float* bv_l = bv + (size_t)l * VIT_D;
        const float* by_l = by + (size_t)l * VIT_D;
        const float* m1w_l = m1w + (size_t)l * VIT_D * VIT_FF;
        const float* m1b_l = m1b + (size_t)l * VIT_FF;
        const float* m2w_l = m2w + (size_t)l * VIT_FF * VIT_D;
        const float* m2b_l = m2b + (size_t)l * VIT_D;

        layernorm_k<<<dim3(NROWS), dim3(256), 0, stream>>>(h, l1w, l1b, z, VIT_D);

        launch_trcvt(stream, wq_l, wT, VIT_D, VIT_D, VIT_D, VIT_D);
        launch_gemm<0, 0>(stream, z, wT, bq_l, nullptr, qb,
                          NROWS, VIT_D, VIT_D, VIT_D, VIT_D, VIT_D, 1, 1, 0, 0, 0, 0, 0, 0);
        launch_trcvt(stream, wk_l, wT, VIT_D, VIT_D, VIT_D, VIT_D);
        launch_gemm<0, 0>(stream, z, wT, bk_l, nullptr, kb,
                          NROWS, VIT_D, VIT_D, VIT_D, VIT_D, VIT_D, 1, 1, 0, 0, 0, 0, 0, 0);
        launch_trcvt(stream, wv_l, wT, VIT_D, VIT_D, VIT_D, VIT_D);
        launch_gemm<0, 0>(stream, z, wT, bv_l, nullptr, vb,
                          NROWS, VIT_D, VIT_D, VIT_D, VIT_D, VIT_D, 1, 1, 0, 0, 0, 0, 0, 0);

        cvt_bf16_k<<<dim3((unsigned)(((size_t)NROWS * VIT_D / 4 + 255) / 256)), dim3(256), 0, stream>>>(
            kb, kbf, (long long)NROWS * VIT_D);

        launch_gemm<0, 0>(stream, qb, kbf, nullptr, nullptr, probs,
                          VIT_S, VIT_S, VIT_HD, VIT_D, VIT_D, VIT_SP,
                          VIT_B * VIT_NH, VIT_NH,
                          SD, VIT_HD, SD, VIT_HD, (long long)VIT_NH * PP, PP);

        softmax_k<<<dim3(VIT_B * VIT_NH * VIT_S), dim3(256), 0, stream>>>(probs);

        launch_trcvt(stream, vb, vt, VIT_S, VIT_HD, VIT_D, VIT_SP,
                     VIT_B * VIT_NH, VIT_NH,
                     SD, VIT_HD, (long long)VIT_NH * VTH, VTH);

        launch_gemm<0, 0>(stream, probs, vt, nullptr, nullptr, yb,
                          VIT_S, VIT_HD, VIT_S, VIT_SP, VIT_SP, VIT_D,
                          VIT_B * VIT_NH, VIT_NH,
                          (long long)VIT_NH * PP, PP, (long long)VIT_NH * VTH, VTH, SD, VIT_HD);

        launch_trcvt(stream, wy_l, wT, VIT_D, VIT_D, VIT_D, VIT_D);
        launch_gemm<0, 1>(stream, yb, wT, by_l, h, h,
                          NROWS, VIT_D, VIT_D, VIT_D, VIT_D, VIT_D, 1, 1, 0, 0, 0, 0, 0, 0);

        layernorm_k<<<dim3(NROWS), dim3(256), 0, stream>>>(h, l2w, l2b, z, VIT_D);

        launch_trcvt(stream, m1w_l, wT, VIT_D, VIT_FF, VIT_FF, VIT_D);
        launch_gemm<1, 0>(stream, z, wT, m1b_l, nullptr, ff,
                          NROWS, VIT_FF, VIT_D, VIT_D, VIT_D, VIT_FF, 1, 1, 0, 0, 0, 0, 0, 0);

        launch_trcvt(stream, m2w_l, wT, VIT_FF, VIT_D, VIT_D, VIT_FF);
        launch_gemm<0, 1>(stream, ff, wT, m2b_l, h, h,
                          NROWS, VIT_D, VIT_FF, VIT_FF, VIT_FF, VIT_D, 1, 1, 0, 0, 0, 0, 0, 0);
    }

    // ---- head ----
    layernorm_k<<<dim3(VIT_B), dim3(256), 0, stream>>>(h, hln_w, hln_b, clsn, SD);
    launch_trcvt(stream, head_w, wT, VIT_D, VIT_NC, VIT_NC, VIT_D);
    launch_gemm<0, 0>(stream, clsn, wT, head_b, nullptr, out,
                      VIT_B, VIT_NC, VIT_D, VIT_D, VIT_D, VIT_NC, 1, 1, 0, 0, 0, 0, 0, 0);
}
